// TalkingHeadsAttention_8821862826380
// MI455X (gfx1250) — compile-verified, matrix operand order checked by reading, not run on hardware
//
#include <hip/hip_runtime.h>

// ---------------------------------------------------------------------------
// Types for CDNA5 WMMA (wave32)
// ---------------------------------------------------------------------------
typedef __attribute__((ext_vector_type(16))) __bf16 v16bf;
typedef __attribute__((ext_vector_type(8)))  __bf16 v8bf;
typedef __attribute__((ext_vector_type(8)))  float  v8f;

__device__ __forceinline__ __bf16 f2bf(float f) {
  union { float f; unsigned u; } v; v.f = f;
  unsigned r = v.u + 0x7fffu + ((v.u >> 16) & 1u);   // round-to-nearest-even
  unsigned short h = (unsigned short)(r >> 16);
  return __builtin_bit_cast(__bf16, h);
}

// Row reduction over the 16 lanes of a half-wave (wave32)
__device__ __forceinline__ float redmax16(float v) {
  v = fmaxf(v, __shfl_xor(v, 1, 32));
  v = fmaxf(v, __shfl_xor(v, 2, 32));
  v = fmaxf(v, __shfl_xor(v, 4, 32));
  v = fmaxf(v, __shfl_xor(v, 8, 32));
  return v;
}
__device__ __forceinline__ float redsum16(float v) {
  v += __shfl_xor(v, 1, 32);
  v += __shfl_xor(v, 2, 32);
  v += __shfl_xor(v, 4, 32);
  v += __shfl_xor(v, 8, 32);
  return v;
}

#define WMMA_BF16(A, B, C) \
  __builtin_amdgcn_wmma_f32_16x16x32_bf16(false, (A), false, (B), (short)0, (C), false, false)

// CDNA5 16-bit A/B fragment: per lane two contiguous 8-element runs,
//   k = 8*khalf + (0..7)  and  k = 16 + 8*khalf + (0..7)
// With bf16 data each run is one 16-byte load (b128), from global or LDS.
__device__ __forceinline__ v16bf load_frag(const __bf16* __restrict__ base, int khalf) {
  v8bf lo = *(const v8bf*)(base + khalf * 8);
  v8bf hi = *(const v8bf*)(base + 16 + khalf * 8);
  v16bf f;
#pragma unroll
  for (int j = 0; j < 8; ++j) { f[j] = lo[j]; f[j + 8] = hi[j]; }
  return f;
}

// ---------------------------------------------------------------------------
// CDNA5 async global->LDS copy (ASYNCcnt path, ISA 15.18.3 / 08_async_tensor)
// ---------------------------------------------------------------------------
__device__ __forceinline__ unsigned lds_addr32(const void* p) {
  return (unsigned)(unsigned long)(__attribute__((address_space(3))) const char*)p;
}
__device__ __forceinline__ void async_cp16(void* lds_dst, const __bf16* gsrc) {
  unsigned dst = lds_addr32(lds_dst);
  unsigned long src = (unsigned long)gsrc;
  asm volatile("global_load_async_to_lds_b128 %0, %1, off"
               :: "v"(dst), "v"(src) : "memory");
}
__device__ __forceinline__ void wait_async0() {
#if __has_builtin(__builtin_amdgcn_s_wait_asynccnt)
  __builtin_amdgcn_s_wait_asynccnt(0);
#else
  asm volatile("s_wait_asynccnt 0" ::: "memory");
#endif
}

// ---------------------------------------------------------------------------
// One-time f32 -> bf16 conversion kernels
// ---------------------------------------------------------------------------
__global__ __launch_bounds__(256) void k_cvt_plain(
    const float* __restrict__ in, __bf16* __restrict__ out, long n) {
  long i = (long)blockIdx.x * 256 + threadIdx.x;
  if (i < n) out[i] = f2bf(in[i]);
}

// out[c*R + r] = in[r*C + c]   (transpose + convert)
__global__ __launch_bounds__(256) void k_cvt_tr(
    const float* __restrict__ in, __bf16* __restrict__ out, int R, int C) {
  long i = (long)blockIdx.x * 256 + threadIdx.x;
  if (i >= (long)R * C) return;
  int c = (int)(i % C);
  int r = (int)(i / C);
  out[(long)c * R + r] = f2bf(in[(long)r * C + c]);
}

// ---------------------------------------------------------------------------
// bf16 GEMM:  C[M,N] = A[M,K] * Bt[N,K]^T (+ bias)
// 256 threads = 8 waves. 128x64 macro-tile/block; each wave owns 32x32 (4 WMMA
// accumulators). K stepped by 32 with double-buffered LDS tiles staged through
// global_load_async_to_lds_b128 (ASYNCcnt) and read back as ds_load_b128 frags.
// ---------------------------------------------------------------------------
template <bool HAS_BIAS>
__global__ __launch_bounds__(256) void k_gemm_bf16(
    const __bf16* __restrict__ A, const __bf16* __restrict__ Bt,
    float* __restrict__ C, const float* __restrict__ bias,
    int M, int N, int K) {
  __shared__ __bf16 sA[2][128 * 32];   // 2 x 8 KB
  __shared__ __bf16 sB[2][64 * 32];    // 2 x 4 KB

  int tid   = threadIdx.x;
  int lane  = tid & 31;
  int wave  = tid >> 5;
  int nb    = N >> 6;
  int tm0   = (blockIdx.x / nb) * 128;
  int tn0   = (blockIdx.x % nb) * 64;
  int wm    = wave & 3;                // 0..3 : 32-row strip
  int wn    = wave >> 2;               // 0..1 : 32-col strip
  int l15   = lane & 15;
  int khalf = lane >> 4;

  // Stage one 128x32 A tile + 64x32 B tile (bf16) into LDS buffer `buf`.
  auto stage = [&](int buf, int k0) {
    // A: 512 chunks of 16B; each thread issues 2
#pragma unroll
    for (int c = tid; c < 512; c += 256) {
      int row = c >> 2, cq = c & 3;
      async_cp16(&sA[buf][row * 32 + cq * 8],
                 A + (long)(tm0 + row) * K + k0 + cq * 8);
    }
    // B: 256 chunks of 16B; one per thread
    {
      int row = tid >> 2, cq = tid & 3;
      async_cp16(&sB[buf][row * 32 + cq * 8],
                 Bt + (long)(tn0 + row) * K + k0 + cq * 8);
    }
  };

  v8f acc[2][2] = {};
  int nk = K >> 5;
  stage(0, 0);
  for (int ik = 0; ik < nk; ++ik) {
    int buf = ik & 1;
    wait_async0();        // own wave's async copies landed in LDS
    __syncthreads();      // => all waves' copies landed
    if (ik + 1 < nk) stage(buf ^ 1, (ik + 1) << 5);

    v16bf a0 = load_frag(&sA[buf][(wm * 32 + l15) * 32], khalf);
    v16bf a1 = load_frag(&sA[buf][(wm * 32 + 16 + l15) * 32], khalf);
    v16bf b0 = load_frag(&sB[buf][(wn * 32 + l15) * 32], khalf);
    v16bf b1 = load_frag(&sB[buf][(wn * 32 + 16 + l15) * 32], khalf);
    acc[0][0] = WMMA_BF16(a0, b0, acc[0][0]);
    acc[0][1] = WMMA_BF16(a0, b1, acc[0][1]);
    acc[1][0] = WMMA_BF16(a1, b0, acc[1][0]);
    acc[1][1] = WMMA_BF16(a1, b1, acc[1][1]);
  }

#pragma unroll
  for (int rm = 0; rm < 2; ++rm) {
#pragma unroll
    for (int cn = 0; cn < 2; ++cn) {
      int col = tn0 + wn * 32 + cn * 16 + l15;
      float bv = HAS_BIAS ? bias[col] : 0.0f;
#pragma unroll
      for (int r = 0; r < 8; ++r) {
        int row = tm0 + wm * 32 + rm * 16 + r + 8 * khalf;
        C[(long)row * N + col] = acc[rm][cn][r] + bv;
      }
    }
  }
}

// ---------------------------------------------------------------------------
// RoPE on q,k + repack from qkv[B,T,3E] (f32) into bf16:
//   Qb,Kb: [B][H][T][D]   Vt: [B][H][D][T]  (transposed for k-contiguous B frags)
// ---------------------------------------------------------------------------
__global__ __launch_bounds__(256) void k_rope_pack(
    const float* __restrict__ qkv,
    __bf16* __restrict__ Qb, __bf16* __restrict__ Kb, __bf16* __restrict__ Vt) {
  long idx = (long)blockIdx.x * 256 + threadIdx.x;  // B*T*H*32 = 2097152
  int dh = idx & 31;
  long r  = idx >> 5;
  int h  = r & 15;  r >>= 4;
  int t  = r & 2047; r >>= 11;
  int b  = (int)r;
  if (b > 1) return;

  long base = ((long)(b * 2048 + t)) * 3072 + h * 64;
  float q1 = qkv[base + dh],        q2 = qkv[base + dh + 32];
  float k1 = qkv[base + 1024 + dh], k2 = qkv[base + 1024 + dh + 32];
  float v1 = qkv[base + 2048 + dh], v2 = qkv[base + 2048 + dh + 32];

  float invf = __expf(-(float)dh * (9.210340371976184f / 32.0f));
  float s, c;
  __sincosf((float)t * invf, &s, &c);

  long ob = (((long)b * 16 + h) * 2048 + t) * 64;
  Qb[ob + dh]      = f2bf(q1 * c - q2 * s);
  Qb[ob + dh + 32] = f2bf(q1 * s + q2 * c);
  Kb[ob + dh]      = f2bf(k1 * c - k2 * s);
  Kb[ob + dh + 32] = f2bf(k1 * s + k2 * c);

  long vb = (((long)b * 16 + h) * 64) * 2048 + t;   // [d][t]
  Vt[vb + (long)dh * 2048]        = f2bf(v1);
  Vt[vb + (long)(dh + 32) * 2048] = f2bf(v2);
}

// ---------------------------------------------------------------------------
// Pass 1: exact softmax stats (m,l) of the PRE-mixed scores.
// Block = 16 waves (one per head) for one (batch, 16-query tile).
// ---------------------------------------------------------------------------
__global__ __launch_bounds__(512) void k_attn_stats(
    const __bf16* __restrict__ Q, const __bf16* __restrict__ K,
    const float* __restrict__ w_pre,
    float* __restrict__ stat_m, float* __restrict__ stat_l) {
  __shared__ float sW[256];
  __shared__ float sS[16 * 256];  // [head][lane*8 + r]

  int tid  = threadIdx.x;
  int lane = tid & 31;
  int h    = tid >> 5;
  int b    = blockIdx.x >> 7;
  int i0   = (blockIdx.x & 127) * 16;
  int l15  = lane & 15;
  int khalf = lane >> 4;

  if (tid < 256) sW[tid] = w_pre[tid] * 0.125f;  // fold scale = D^-0.5
  __syncthreads();

  const __bf16* Qh = Q + ((long)b * 16 + h) * 2048 * 64;
  const __bf16* Kh = K + ((long)b * 16 + h) * 2048 * 64;

  const __bf16* qrow = Qh + (long)(i0 + l15) * 64;
  v16bf a0 = load_frag(qrow, khalf);
  v16bf a1 = load_frag(qrow + 32, khalf);

  float m[8], l[8];
#pragma unroll
  for (int r = 0; r < 8; ++r) { m[r] = -3.0e38f; l[r] = 0.0f; }

  for (int j0 = 0; j0 < 2048; j0 += 16) {
    const __bf16* krow = Kh + (long)(j0 + l15) * 64;
    v16bf b0 = load_frag(krow, khalf);
    v16bf b1 = load_frag(krow + 32, khalf);
    v8f s = {};
    s = WMMA_BF16(a0, b0, s);
    s = WMMA_BF16(a1, b1, s);
#pragma unroll
    for (int r = 0; r < 8; ++r) sS[h * 256 + lane * 8 + r] = s[r];
    __syncthreads();

#pragma unroll
    for (int r = 0; r < 8; ++r) {
      float pre = 0.0f;
#pragma unroll
      for (int hh = 0; hh < 16; ++hh)
        pre = __builtin_fmaf(sW[h * 16 + hh], sS[hh * 256 + lane * 8 + r], pre);
      float mnew = fmaxf(m[r], redmax16(pre));
      float e    = __expf(pre - mnew);
      float esum = redsum16(e);
      l[r] = l[r] * __expf(m[r] - mnew) + esum;
      m[r] = mnew;
    }
    __syncthreads();
  }

  if (l15 == 0) {
#pragma unroll
    for (int r = 0; r < 8; ++r) {
      int row = r + 8 * khalf;
      long si = ((long)b * 16 + h) * 2048 + i0 + row;
      stat_m[si] = m[r];
      stat_l[si] = l[r];
    }
  }
}

// ---------------------------------------------------------------------------
// Pass 2: recompute scores, exact-normalize with (m,l), pre-mix -> exp ->
// post-mix -> P.V (WMMA), write ctx (bf16) in [B,T,E] layout.
// ---------------------------------------------------------------------------
__global__ __launch_bounds__(512) void k_attn_out(
    const __bf16* __restrict__ Q, const __bf16* __restrict__ K,
    const __bf16* __restrict__ Vt,
    const float* __restrict__ w_pre, const float* __restrict__ w_post,
    const float* __restrict__ stat_m, const float* __restrict__ stat_l,
    __bf16* __restrict__ ctx) {
  __shared__ float sWpre[256];
  __shared__ float sWpost[256];
  __shared__ float sS[16 * 512];  // [head][ctile*256 + lane*8 + r]
  __shared__ float sP[16 * 512];

  int tid   = threadIdx.x;
  int lane  = tid & 31;
  int g     = tid >> 5;
  int b     = blockIdx.x >> 7;
  int i0    = (blockIdx.x & 127) * 16;
  int l15   = lane & 15;
  int khalf = lane >> 4;

  if (tid < 256) { sWpre[tid] = w_pre[tid] * 0.125f; sWpost[tid] = w_post[tid]; }
  __syncthreads();

  const __bf16* Qh = Q  + ((long)b * 16 + g) * 2048 * 64;
  const __bf16* Kh = K  + ((long)b * 16 + g) * 2048 * 64;
  const __bf16* Vh = Vt + ((long)b * 16 + g) * 64 * 2048;  // [d][t]

  const __bf16* qrow = Qh + (long)(i0 + l15) * 64;
  v16bf a0 = load_frag(qrow, khalf);
  v16bf a1 = load_frag(qrow + 32, khalf);

  float m[8], rl[8];
#pragma unroll
  for (int r = 0; r < 8; ++r) {
    long si = ((long)b * 16 + g) * 2048 + i0 + r + 8 * khalf;
    m[r]  = stat_m[si];
    rl[r] = 1.0f / stat_l[si];
  }

  v8f o[4] = {};  // 16 rows x 64 dims as four 16x16 tiles

  for (int j0 = 0; j0 < 2048; j0 += 32) {
    // --- raw scores S_g for 32 keys (two 16-col C tiles) ---
#pragma unroll
    for (int c = 0; c < 2; ++c) {
      const __bf16* krow = Kh + (long)(j0 + c * 16 + l15) * 64;
      v16bf b0 = load_frag(krow, khalf);
      v16bf b1 = load_frag(krow + 32, khalf);
      v8f s = {};
      s = WMMA_BF16(a0, b0, s);
      s = WMMA_BF16(a1, b1, s);
#pragma unroll
      for (int r = 0; r < 8; ++r) sS[g * 512 + c * 256 + lane * 8 + r] = s[r];
    }
    __syncthreads();

    // --- pre-mix + exact softmax -> per-head probabilities P_g ---
#pragma unroll
    for (int c = 0; c < 2; ++c) {
#pragma unroll
      for (int r = 0; r < 8; ++r) {
        float pre = 0.0f;
#pragma unroll
        for (int hh = 0; hh < 16; ++hh)
          pre = __builtin_fmaf(sWpre[g * 16 + hh],
                               sS[hh * 512 + c * 256 + lane * 8 + r], pre);
        sP[g * 512 + c * 256 + lane * 8 + r] = __expf(pre - m[r]) * rl[r];
      }
    }
    __syncthreads();

    // --- post-mix -> attn_post_g, stored back into sS[g] (own region) ---
#pragma unroll
    for (int c = 0; c < 2; ++c) {
#pragma unroll
      for (int r = 0; r < 8; ++r) {
        float pp = 0.0f;
#pragma unroll
        for (int hh = 0; hh < 16; ++hh)
          pp = __builtin_fmaf(sWpost[g * 16 + hh],
                              sP[hh * 512 + c * 256 + lane * 8 + r], pp);
        sS[g * 512 + c * 256 + lane * 8 + r] = pp;
      }
    }

    // --- gather attn_post_g as a 16x32 bf16 A fragment (wave-local LDS) ---
    v16bf pa;
#pragma unroll
    for (int j = 0; j < 16; ++j) {
      int k = ((j >> 3) << 4) | (khalf << 3) | (j & 7);  // key column 0..31
      int c = k >> 4, n = k & 15;
      int srclane = n + ((l15 >> 3) << 4);
      int srcr    = l15 & 7;
      pa[j] = f2bf(sS[g * 512 + c * 256 + srclane * 8 + srcr]);
    }

    // --- O_g += P_g(16x32) * V_g(32x64): V^T rows are k-contiguous ---
#pragma unroll
    for (int dc = 0; dc < 4; ++dc) {
      v16bf vb = load_frag(Vh + (long)(dc * 16 + l15) * 2048 + j0, khalf);
      o[dc] = WMMA_BF16(pa, vb, o[dc]);
    }
    __syncthreads();  // protect sS/sP before next iteration's writes
  }

  // ctx[b, i, g*64 + d] in bf16 (feeds final bf16 GEMM directly)
#pragma unroll
  for (int dc = 0; dc < 4; ++dc) {
#pragma unroll
    for (int r = 0; r < 8; ++r) {
      int row = r + 8 * khalf;
      ctx[((long)(b * 2048) + i0 + row) * 1024 + g * 64 + dc * 16 + l15] =
          f2bf(o[dc][r]);
    }
  }
}

// ---------------------------------------------------------------------------
// Launch
// ---------------------------------------------------------------------------
extern "C" void kernel_launch(void* const* d_in, const int* in_sizes, int n_in,
                              void* d_out, int out_size, void* d_ws, size_t ws_size,
                              hipStream_t stream) {
  const float* x      = (const float*)d_in[0];  // [2,2048,1024]
  const float* w_qkv  = (const float*)d_in[1];  // [1024,3072]
  const float* w_pre  = (const float*)d_in[2];  // [16,16]
  const float* w_post = (const float*)d_in[3];  // [16,16]
  const float* w_out  = (const float*)d_in[4];  // [1024,1024]
  const float* b_out  = (const float*)d_in[5];  // [1024]
  float* out = (float*)d_out;

  char* ws = (char*)d_ws;
  float*  qkv   = (float*)ws;                      // 48 MB  (4096x3072 f32)
  __bf16* Xb    = (__bf16*)(ws + 50331648);        //  8 MB  x bf16 [4096][1024]
  __bf16* Wqt   = (__bf16*)(ws + 58720256);        //  6 MB  w_qkv^T bf16 [3072][1024]
  __bf16* Wob   = (__bf16*)(ws + 65011712);        //  2 MB  w_out bf16 [1024][1024]
  __bf16* Qb    = (__bf16*)(ws + 67108864);        //  8 MB
  __bf16* Kb    = (__bf16*)(ws + 75497472);        //  8 MB
  __bf16* Vtb   = (__bf16*)(ws + 83886080);        //  8 MB  (transposed [d][t])
  float*  sm    = (float*)(ws + 92274688);         // 256 KB
  float*  sl    = (float*)(ws + 92536832);         // 256 KB
  __bf16* ctxb  = (__bf16*)(ws + 92798976);        //  8 MB

  // 0) one-time bf16 conversions
  k_cvt_plain<<<16384, 256, 0, stream>>>(x, Xb, 4194304L);
  k_cvt_tr<<<12288, 256, 0, stream>>>(w_qkv, Wqt, 1024, 3072);
  k_cvt_plain<<<4096, 256, 0, stream>>>(w_out, Wob, 1048576L);
  // 1) qkv = x @ w_qkv      (128x64 macro-tiles: 32 x 48 blocks)
  k_gemm_bf16<false><<<1536, 256, 0, stream>>>(Xb, Wqt, qkv, nullptr, 4096, 3072, 1024);
  // 2) RoPE + repack (V transposed)
  k_rope_pack<<<8192, 256, 0, stream>>>(qkv, Qb, Kb, Vtb);
  // 3) pass 1: pre-mixed softmax stats
  k_attn_stats<<<256, 512, 0, stream>>>(Qb, Kb, w_pre, sm, sl);
  // 4) pass 2: talking-heads attention -> ctx (bf16)
  k_attn_out<<<256, 512, 0, stream>>>(Qb, Kb, Vtb, w_pre, w_post, sm, sl, ctxb);
  // 5) out = ctx @ w_out^T + b_out   (32 x 16 blocks)
  k_gemm_bf16<true><<<512, 256, 0, stream>>>(ctxb, Wob, out, b_out, 4096, 1024, 1024);
}
